// SpectralAttentionLayer_21242908246138
// MI455X (gfx1250) — matrix operand order repaired and checked by reading, not compiled, not run on hardware
//
#include <hip/hip_runtime.h>
#include <hip/hip_bf16.h>
#include <math.h>

// ---------------------------------------------------------------------------
// SpectralAttentionLayer for MI455X (gfx1250, wave32, WMMA)
//
//   D_MODEL=1024, H=16, Dh=64, K_EIG=16, B=4, N=4096 -> 16384 tokens
//
// Algebra: Qf = (x@Wq) folded with eigenvectors+filter  ==  x @ Wq'
//   where Wq'[c, k*64+d] = w[k] * sum_h E[h,k] * Wq[c, h*64+d]   (same K,V)
// Then per token: scores = Qf@Kf^T/8, softmax(16), out_f = attn@Vf,
//                 out = E @ out_f.
// Projections: bf16 hi/lo split (3x v_wmma_f32_16x16x32_bf16) ~ fp32 accuracy.
// Attention:   v_wmma_f32_16x16x4_f32 (pure fp32).
// ---------------------------------------------------------------------------

typedef __attribute__((ext_vector_type(16))) __bf16 v16bf;
typedef __attribute__((ext_vector_type(8)))  float  v8f;
typedef __attribute__((ext_vector_type(2)))  float  v2f;

// Workspace layout (in dwords):
//  [0, WB_DWORDS)          : swizzled bf16 hi/lo folded weights
//      dword idx = (((ct*32 + kc)*32 + lane)*16 + slot)
//      ct = mat*64 + colTile (0..191), kc = K-chunk (0..31), lane 0..31,
//      slot 0..7 = hi dwords (v16bf), slot 8..15 = lo dwords (v16bf)
//      strides (bytes): ct -> 65536, kc -> 2048, lane -> 64
//  [WB_DWORDS, +3*1024)    : folded biases (f32), per matrix 1024 cols
static constexpr unsigned WB_DWORDS = 3u * 64u * 32u * 32u * 16u; // 12 MB

// 16-bit B-matrix 32x16 K index: lane half g holds K=8g..8g+7 and 16+8g..16+8g+7
__device__ __forceinline__ int klocalB(int g, int v, int t) {
  return (v >> 2) * 16 + g * 8 + (v & 3) * 2 + t;
}

// ---------------------------------------------------------------------------
// Kernel 1: fold eigenvectors+filter into weights, split bf16 hi/lo, swizzle
// into WMMA-B register order. One thread per output dword pair (hi,lo).
// idx = mat<<19 | ct<<13 | kc<<8 | lane<<3 | v   (3<<19 threads total)
// ---------------------------------------------------------------------------
__global__ void fold_weights(const float* __restrict__ Wq,
                             const float* __restrict__ Wk,
                             const float* __restrict__ Wv,
                             const float* __restrict__ E,
                             const float* __restrict__ fw,
                             float* __restrict__ ws) {
  unsigned idx = blockIdx.x * blockDim.x + threadIdx.x;
  if (idx >= (3u << 19)) return;
  int v    = idx & 7;
  int lane = (idx >> 3) & 31;
  int kc   = (idx >> 8) & 31;
  int ct   = (idx >> 13) & 63;
  int mat  = idx >> 19;
  const float* W = (mat == 0) ? Wq : ((mat == 1) ? Wk : Wv);

  int n = lane & 15, g = lane >> 4;
  int col = ct * 16 + n;          // 0..1023 within this matrix
  int ke = col >> 6, d = col & 63;
  float scale = fw[ke];

  unsigned dw_hi = 0, dw_lo = 0;
#pragma unroll
  for (int t = 0; t < 2; ++t) {
    int c = kc * 32 + klocalB(g, v, t);   // K dim of GEMM = model dim c
    float acc = 0.f;
#pragma unroll
    for (int h = 0; h < 16; ++h)
      acc += E[h * 16 + ke] * W[c * 1024 + h * 64 + d];
    acc *= scale;
    __bf16 hi = (__bf16)acc;
    __bf16 lo = (__bf16)(acc - (float)hi);
    dw_hi |= ((unsigned)__builtin_bit_cast(unsigned short, hi)) << (16 * t);
    dw_lo |= ((unsigned)__builtin_bit_cast(unsigned short, lo)) << (16 * t);
  }
  unsigned ctg  = (unsigned)(mat * 64 + ct);
  unsigned base = ((ctg * 32u + (unsigned)kc) * 32u + (unsigned)lane) * 16u;
  unsigned* wsu = (unsigned*)ws;
  wsu[base + v]     = dw_hi;
  wsu[base + 8 + v] = dw_lo;
}

// Fold biases (zero in practice, but kept for correctness of the mapping).
__global__ void fold_bias(const float* __restrict__ bq,
                          const float* __restrict__ bk,
                          const float* __restrict__ bv,
                          const float* __restrict__ E,
                          const float* __restrict__ fw,
                          float* __restrict__ ws) {
  int idx = blockIdx.x * blockDim.x + threadIdx.x;
  if (idx >= 3 * 1024) return;
  int mat = idx >> 10, col = idx & 1023;
  const float* b = (mat == 0) ? bq : ((mat == 1) ? bk : bv);
  int ke = col >> 6, d = col & 63;
  float acc = 0.f;
#pragma unroll
  for (int h = 0; h < 16; ++h) acc += E[h * 16 + ke] * b[h * 64 + d];
  ws[WB_DWORDS + idx] = acc * fw[ke];
}

// ---------------------------------------------------------------------------
// Kernel 2: fused projections + spectral attention.
// Block = 512 threads (16 waves), owns 16 tokens. Each wave computes 12 of
// 192 col-tiles (3 matrices x 64 tiles) with split-bf16 WMMA, stages
// Qf/Kf/Vf in LDS, then wave w runs token w's attention with fp32 WMMA.
// ---------------------------------------------------------------------------
__global__ void __launch_bounds__(512, 1)
spectral_attn(const float* __restrict__ x,
              const float* __restrict__ ws,
              const float* __restrict__ E,
              float* __restrict__ out) {
  __shared__ float ldsQ[16 * 1024];   // Qf per token, later reused for out_f
  __shared__ float ldsK[16 * 1024];   // Kf per token
  __shared__ float ldsV[16 * 1024];   // Vf per token
  __shared__ float ldsS[16 * 256];    // per-wave 16x16 scores/attn scratch

  const int tok0 = blockIdx.x * 16;           // 1024 blocks x 16 tokens
  const int wave = threadIdx.x >> 5;
  const int lane = threadIdx.x & 31;
  const int n = lane & 15, g = lane >> 4;

  // ---------------- Phase 1: C[16 tokens x 192 tiles] = x @ W' -------------
  v8f acc[12];
#pragma unroll
  for (int i = 0; i < 12; ++i) {
    int ct = wave * 12 + i;
    int mat = ct / 64, c16 = ct % 64;
    float b = ws[WB_DWORDS + mat * 1024 + c16 * 16 + n];
#pragma unroll
    for (int j = 0; j < 8; ++j) acc[i][j] = b;
  }

  const float* xrow = x + (size_t)(tok0 + n) * 1024;  // A row for this lane

  // Single base pointer for this wave's 12 B tiles; all tile/buffer offsets
  // are compile-time constants that fit the signed 24-bit VMEM offset, so the
  // inner loop has zero address VALU (one pointer bump per K-chunk).
  const char* bptr = (const char*)ws
                   + (size_t)(wave * 12) * 65536u   // ct stride
                   + (size_t)lane * 64u;            // lane stride

  for (int kc = 0; kc < 32; ++kc) {
    // Load & convert A chunk (16x32 f32 -> bf16 hi/lo) in A-register layout:
    // per lane two contiguous 8-float runs: [kc*32+g*8, +8) and [+16+g*8, +8)
    v16bf ahi, alo;
    const int b1 = kc * 32 + g * 8;
    const int b2 = kc * 32 + 16 + g * 8;
#pragma unroll
    for (int j = 0; j < 8; ++j) {
      float f1 = xrow[b1 + j];
      __bf16 h1 = (__bf16)f1;
      ahi[j] = h1;  alo[j] = (__bf16)(f1 - (float)h1);
      float f2 = xrow[b2 + j];
      __bf16 h2 = (__bf16)f2;
      ahi[8 + j] = h2;  alo[8 + j] = (__bf16)(f2 - (float)h2);
    }
#pragma unroll
    for (int i = 0; i < 12; ++i) {
      v16bf bhi = *(const v16bf*)(bptr + i * 65536);        // offset:imm
      v16bf blo = *(const v16bf*)(bptr + i * 65536 + 32);   // offset:imm+32
      // (ahi+alo)*(bhi+blo) ~= ahi*bhi + ahi*blo + alo*bhi  (fp32-accurate)
      acc[i] = __builtin_amdgcn_wmma_f32_16x16x32_bf16(false, ahi, false, bhi,
                                                       (short)0, acc[i], false, false);
      acc[i] = __builtin_amdgcn_wmma_f32_16x16x32_bf16(false, ahi, false, blo,
                                                       (short)0, acc[i], false, false);
      acc[i] = __builtin_amdgcn_wmma_f32_16x16x32_bf16(false, alo, false, bhi,
                                                       (short)0, acc[i], false, false);
    }
    bptr += 2048;                                           // kc stride
  }

  // Scatter C tiles to LDS: C layout -> VGPR r: M = r + 8*g (token), N = n.
#pragma unroll
  for (int i = 0; i < 12; ++i) {
    int ct = wave * 12 + i;
    int mat = ct / 64, c16 = ct % 64;
    float* dst = (mat == 0) ? ldsQ : ((mat == 1) ? ldsK : ldsV);
#pragma unroll
    for (int r = 0; r < 8; ++r)
      dst[(r + 8 * g) * 1024 + c16 * 16 + n] = acc[i][r];
  }
  __syncthreads();

  // ---------------- Phase 2: per-token attention (wave w -> token w) -------
  {
    const int tok = wave;
    const float* Qf = ldsQ + tok * 1024;    // [k_eig 16][d 64]
    const float* Kf = ldsK + tok * 1024;
    const float* Vf = ldsV + tok * 1024;
    float* S = ldsS + tok * 256;            // 16x16 scores/attn

    // scores = Qf @ Kf^T   (M=k, N=l, K=d=64) in 16 fp32-WMMA steps
    v8f sc = {};
#pragma unroll
    for (int kc = 0; kc < 16; ++kc) {
      int kd = kc * 4 + g * 2;              // f32 A/B: K = g*2 + vgpr
      v2f a;  a[0] = Qf[n * 64 + kd];  a[1] = Qf[n * 64 + kd + 1];
      v2f b;  b[0] = Kf[n * 64 + kd];  b[1] = Kf[n * 64 + kd + 1];
      sc = __builtin_amdgcn_wmma_f32_16x16x4_f32(false, a, false, b,
                                                 (short)0, sc, false, false);
    }
#pragma unroll
    for (int r = 0; r < 8; ++r) S[(r + 8 * g) * 16 + n] = sc[r];
    asm volatile("s_wait_dscnt 0x0" ::: "memory");

    // softmax over l (row-wise); lanes 0..15 each own one row
    if (lane < 16) {
      float row[16];
      float mx = -3.402823e38f;
#pragma unroll
      for (int l = 0; l < 16; ++l) {
        float v = S[lane * 16 + l] * 0.125f;  // 1/sqrt(Dh)
        row[l] = v;
        mx = fmaxf(mx, v);
      }
      float sum = 0.f;
#pragma unroll
      for (int l = 0; l < 16; ++l) { float e = __expf(row[l] - mx); row[l] = e; sum += e; }
      float inv = 1.f / sum;
#pragma unroll
      for (int l = 0; l < 16; ++l) S[lane * 16 + l] = row[l] * inv;
    }
    asm volatile("s_wait_dscnt 0x0" ::: "memory");

    // out_f = attn @ Vf   (M=k, N=d tile of 16, K=l=16)
    float* OF = ldsQ + tok * 1024;          // reuse Qf region: [k 16][d 64]
    v8f of[4];
#pragma unroll
    for (int dt = 0; dt < 4; ++dt) {
      v8f o = {};
#pragma unroll
      for (int kc = 0; kc < 4; ++kc) {
        int kl = kc * 4 + g * 2;
        v2f a;  a[0] = S[n * 16 + kl];       a[1] = S[n * 16 + kl + 1];
        v2f b;  b[0] = Vf[kl * 64 + dt * 16 + n];
                b[1] = Vf[(kl + 1) * 64 + dt * 16 + n];
        o = __builtin_amdgcn_wmma_f32_16x16x4_f32(false, a, false, b,
                                                  (short)0, o, false, false);
      }
      of[dt] = o;
    }
#pragma unroll
    for (int dt = 0; dt < 4; ++dt)
#pragma unroll
      for (int r = 0; r < 8; ++r)
        OF[(r + 8 * g) * 64 + dt * 16 + n] = of[dt][r];
    asm volatile("s_wait_dscnt 0x0" ::: "memory");

    // out = E @ out_f   (M=h, N=d tile, K=k=16), store straight to global
    float* orow = out + (size_t)(tok0 + tok) * 1024;
#pragma unroll
    for (int dt = 0; dt < 4; ++dt) {
      v8f o = {};
#pragma unroll
      for (int kc = 0; kc < 4; ++kc) {
        int kk = kc * 4 + g * 2;
        v2f a;  a[0] = E[n * 16 + kk];       a[1] = E[n * 16 + kk + 1];
        v2f b;  b[0] = OF[kk * 64 + dt * 16 + n];
                b[1] = OF[(kk + 1) * 64 + dt * 16 + n];
        o = __builtin_amdgcn_wmma_f32_16x16x4_f32(false, a, false, b,
                                                  (short)0, o, false, false);
      }
#pragma unroll
      for (int r = 0; r < 8; ++r)
        orow[(r + 8 * g) * 64 + dt * 16 + n] = o[r];   // [h=r+8g][d]
    }
  }
}

// ---------------------------------------------------------------------------
extern "C" void kernel_launch(void* const* d_in, const int* in_sizes, int n_in,
                              void* d_out, int out_size, void* d_ws, size_t ws_size,
                              hipStream_t stream) {
  const float* x  = (const float*)d_in[0];
  const float* Wq = (const float*)d_in[1];
  const float* bq = (const float*)d_in[2];
  const float* Wk = (const float*)d_in[3];
  const float* bk = (const float*)d_in[4];
  const float* Wv = (const float*)d_in[5];
  const float* bv = (const float*)d_in[6];
  const float* E  = (const float*)d_in[7];
  const float* fw = (const float*)d_in[8];
  float* out = (float*)d_out;
  float* ws  = (float*)d_ws;   // needs 12 MB + 12 KB

  // 1) fold eigenvectors/filter into weights, bf16 hi/lo, WMMA-B swizzle
  fold_weights<<<(3 << 19) / 256, 256, 0, stream>>>(Wq, Wk, Wv, E, fw, ws);
  fold_bias<<<12, 256, 0, stream>>>(bq, bk, bv, E, fw, ws);
  // 2) fused projection GEMM + per-token spectral attention
  spectral_attn<<<(4 * 4096) / 16, 512, 0, stream>>>(x, ws, E, out);
}
